// MGRUCell_89567247991384
// MI455X (gfx1250) — matrix operand order repaired and checked by reading, not compiled, not run on hardware
//
#include <hip/hip_runtime.h>
#include <hip/hip_bf16.h>

typedef __attribute__((ext_vector_type(16))) _Float16 v16h;
typedef __attribute__((ext_vector_type(8)))  float    v8f;
typedef __attribute__((ext_vector_type(4)))  float    v4f;

#define BB 256
#define II 512
#define HH 512
#define OO 512
#define KK 256
#define BH (BB*HH)          // 131072
#define KCHUNK 64           // K split: 4 chunks of 64 -> 4x wave concurrency

// -------------------- helpers --------------------

__device__ __forceinline__ float sigmoidf_fast(float x) {
    return 1.0f / (1.0f + __expf(-x));     // exp(-x)->inf gives 0, ok
}
__device__ __forceinline__ float tanhf_fast(float x) {
    float xc = fminf(fmaxf(x, -15.0f), 15.0f);
    float e  = __expf(2.0f * xc);
    return (e - 1.0f) / (e + 1.0f);
}

// -------------------- f32 -> f16 converters / packers --------------------

__global__ __launch_bounds__(256) void convert_f32_f16_v4(
    const float* __restrict__ s, _Float16* __restrict__ d, int n4)
{
    int i = blockIdx.x * blockDim.x + threadIdx.x;
    if (i < n4) {
        v4f v = ((const v4f*)s)[i];
        _Float16 h[4] = {(_Float16)v.x, (_Float16)v.y, (_Float16)v.z, (_Float16)v.w};
        ((uint2*)d)[i] = *(const uint2*)h;
    }
}

// builds Xd = f16([sample|hidden|d0]) [256,1536] and Xc = f16([sample|hidden]) [256,1024]
__global__ __launch_bounds__(256) void pack_inputs(
    const float* __restrict__ sample, const float* __restrict__ hidden,
    const float* __restrict__ d0, _Float16* __restrict__ Xd, _Float16* __restrict__ Xc)
{
    int col = blockIdx.x * 256 + threadIdx.x;   // 0..1535  (grid.x = 6)
    int row = blockIdx.y;                       // 0..255
    float v;
    if      (col < 512)  v = sample[row * 512 + col];
    else if (col < 1024) v = hidden[row * 512 + (col - 512)];
    else                 v = d0[row * 512 + (col - 1024)];
    _Float16 h = (_Float16)v;
    Xd[row * 1536 + col] = h;
    if (col < 1024) Xc[row * 1024 + col] = h;
}

// builds X2 = f16([sample | r_gate*hidden]) [256,1024]
__global__ __launch_bounds__(256) void pack_combined2(
    const float* __restrict__ sample, const float* __restrict__ hidden,
    const float* __restrict__ r, _Float16* __restrict__ X2)
{
    int col = blockIdx.x * 256 + threadIdx.x;   // 0..1023  (grid.x = 4)
    int row = blockIdx.y;                       // 0..255
    float v;
    if (col < 512) v = sample[row * 512 + col];
    else {
        int c = col - 512;
        v = r[row * 512 + c] * hidden[row * 512 + c];
    }
    X2[row * 1024 + col] = (_Float16)v;
}

// -------------------- WMMA GEMM:  Y[256,512] = epi(X[256,Kd] @ W[512,Kd]^T + b) --------------------
// A frag (16x32 f16): lane m = lane&15, half = lane>>4;
//   VGPR v<4 : k = half*8 + 2v          -> dwords [half*4 + v]
//   VGPR v>=4: k = 16 + half*8 + 2(v-4) -> dwords [8 + half*4 + (v-4)]
// B frag (32x16 f16): lane n = lane&15, half = lane>>4; same k layout as A (dwords [half*4 + v], [8 + half*4 + (v-4)])
// C/D (16x16 f32): lane n = lane&15; VGPR r holds row m = r + 8*(lane>>4)

union Frag { uint4 q[2]; unsigned int u[8]; v16h h; };

template <int EPI>   // 0: sigmoid*0.5   1: sigmoid   2: tanh*(1-z)   3: identity
__global__ __launch_bounds__(256) void gemm_wmma(
    const _Float16* __restrict__ X, const _Float16* __restrict__ W,
    const float* __restrict__ bias, int Kd,
    float* __restrict__ Y, const float* __restrict__ zgate)
{
    const int lane  = threadIdx.x & 31;
    const int wave  = threadIdx.x >> 5;
    const int mTile = blockIdx.x >> 2;                    // 16 M tiles
    const int nTile = ((blockIdx.x & 3) << 3) + wave;     // 32 N tiles
    const int m0 = mTile * 16, n0 = nTile * 16;
    const int half = lane >> 4;
    const int mr   = lane & 15;

    const _Float16* Xrow = X + (size_t)(m0 + mr) * Kd;    // A row m
    const _Float16* Wrow = W + (size_t)(n0 + mr) * Kd;    // B column n (W row-major [H,Kd])

    v8f acc = {0.f, 0.f, 0.f, 0.f, 0.f, 0.f, 0.f, 0.f};

    for (int kk = 0; kk < Kd; kk += 32) {
        Frag a, b;
        const uint4* xa = (const uint4*)(Xrow + kk);
        const uint4* wb = (const uint4*)(Wrow + kk);
        a.q[0] = xa[half];
        a.q[1] = xa[2 + half];
        b.q[0] = wb[half];
        b.q[1] = wb[2 + half];
        acc = __builtin_amdgcn_wmma_f32_16x16x32_f16(
            false, a.h, false, b.h, (short)0, acc, false, false);
    }

    const int col = n0 + mr;
    const float bv = bias[col];
#pragma unroll
    for (int r = 0; r < 8; ++r) {
        const int row = m0 + r + half * 8;
        float x = acc[r] + bv;
        float y;
        if      (EPI == 0) y = sigmoidf_fast(x) * 0.5f;
        else if (EPI == 1) y = sigmoidf_fast(x);
        else if (EPI == 2) y = tanhf_fast(x) * (1.0f - zgate[row * HH + col]);
        else               y = x;
        Y[row * HH + col] = y;
    }
}

// -------------------- fractional filter, K-split, fused h_c_1 shift --------------------
// first = -sum_{m=0..K-1} ht[K-1-m] * p_m,  p_m = prod_{t<=m} (t-d)/(t+1).
// Chunk c handles m in [64c, 64c+64): recompute prefix product (pure VALU, hidden
// under memory latency), stream its slices once (NT), and write each slice j=255-m
// to h_c_1[j-1] (NT) -- fusing the state shift saves a full 134 MB re-read.
__global__ __launch_bounds__(256) void filter_partial(
    const float* __restrict__ ht, const float* __restrict__ dvals,
    float* __restrict__ Spart,               // [4][BH] partial sums
    float* __restrict__ hc1)                 // h_c_1 base [KK][BH]
{
    __shared__ float inv[KK];
    const int t = threadIdx.x;
    inv[t] = 1.0f / (float)(t + 1);
    __syncthreads();

    const int chunk = blockIdx.y;                       // 0..3
    const int base  = (blockIdx.x * 256 + t) * 4;       // 128 blocks cover BH
    const v4f d = *(const v4f*)(dvals + base);

    float p0 = 1.f, p1 = 1.f, p2 = 1.f, p3 = 1.f;
    const int mstart = chunk * KCHUNK;
    for (int m2 = 0; m2 < mstart; ++m2) {               // prefix product, no loads
        const float r = inv[m2], mf = (float)m2;
        p0 *= (mf - d.x) * r;
        p1 *= (mf - d.y) * r;
        p2 *= (mf - d.z) * r;
        p3 *= (mf - d.w) * r;
    }

    float s0 = 0.f, s1 = 0.f, s2 = 0.f, s3 = 0.f;
    for (int m = mstart; m < mstart + KCHUNK; ++m) {
        const float r = inv[m], mf = (float)m;
        p0 *= (mf - d.x) * r;
        p1 *= (mf - d.y) * r;
        p2 *= (mf - d.z) * r;
        p3 *= (mf - d.w) * r;
        const int j = KK - 1 - m;
        const v4f h = __builtin_nontemporal_load((const v4f*)(ht + (size_t)j * BH + base));
        s0 += h.x * p0;
        s1 += h.y * p1;
        s2 += h.z * p2;
        s3 += h.w * p3;
        if (j >= 1)                                     // shift: h_c_1[j-1] = ht[j]
            __builtin_nontemporal_store(h, (v4f*)(hc1 + (size_t)(j - 1) * BH + base));
    }
    v4f sv; sv.x = s0; sv.y = s1; sv.z = s2; sv.w = s3;
    *(v4f*)(Spart + (size_t)chunk * BH + base) = sv;
}

// hidden_new = second - (S0+S1+S2+S3); write f32 (x2 dests) + f16 copy for Wo GEMM
__global__ __launch_bounds__(256) void combine_hidden(
    const float* __restrict__ Spart, const float* __restrict__ second,
    float* __restrict__ hidden_out, float* __restrict__ hc1_last,
    _Float16* __restrict__ hnew16)
{
    const int base = (blockIdx.x * 256 + threadIdx.x) * 4;
    v4f s = *(const v4f*)(Spart + base);
    s += *(const v4f*)(Spart + (size_t)BH + base);
    s += *(const v4f*)(Spart + (size_t)2 * BH + base);
    s += *(const v4f*)(Spart + (size_t)3 * BH + base);
    const v4f sec = *(const v4f*)(second + base);
    const v4f hn = sec - s;
    *(v4f*)(hidden_out + base) = hn;
    *(v4f*)(hc1_last + base)   = hn;
    _Float16 h4[4] = {(_Float16)hn.x, (_Float16)hn.y, (_Float16)hn.z, (_Float16)hn.w};
    *(uint2*)(hnew16 + base) = *(const uint2*)h4;
}

// -------------------- launch --------------------

extern "C" void kernel_launch(void* const* d_in, const int* in_sizes, int n_in,
                              void* d_out, int out_size, void* d_ws, size_t ws_size,
                              hipStream_t stream)
{
    (void)in_sizes; (void)n_in; (void)out_size; (void)ws_size;

    const float* sample = (const float*)d_in[0];
    const float* hidden = (const float*)d_in[1];
    const float* htens  = (const float*)d_in[2];
    const float* d0     = (const float*)d_in[3];
    const float* Wd = (const float*)d_in[4];  const float* bd = (const float*)d_in[5];
    const float* Wr = (const float*)d_in[6];  const float* br = (const float*)d_in[7];
    const float* Wz = (const float*)d_in[8];  const float* bz = (const float*)d_in[9];
    const float* Wh = (const float*)d_in[10]; const float* bh = (const float*)d_in[11];
    const float* Wo = (const float*)d_in[12]; const float* bo = (const float*)d_in[13];

    float* out_output = (float*)d_out;                       // [256,512]
    float* out_hidden = out_output + BB * OO;                // [256,512]
    float* out_hc1    = out_hidden + BB * HH;                // [256,256,512]
    float* out_dvals  = out_hc1 + (size_t)KK * BH;           // [256,512]

    // workspace carve-up (256B aligned)
    char* ws = (char*)d_ws;
    size_t off = 0;
    auto take = [&](size_t bytes) { char* p = ws + off; off = (off + bytes + 255) & ~(size_t)255; return p; };
    _Float16* Wd16 = (_Float16*)take((size_t)HH * (II + 2 * HH) * 2);  // 512x1536
    _Float16* Wr16 = (_Float16*)take((size_t)HH * (II + HH) * 2);      // 512x1024
    _Float16* Wz16 = (_Float16*)take((size_t)HH * (II + HH) * 2);
    _Float16* Wh16 = (_Float16*)take((size_t)HH * (II + HH) * 2);
    _Float16* Wo16 = (_Float16*)take((size_t)OO * HH * 2);             // 512x512
    _Float16* Xd16 = (_Float16*)take((size_t)BB * (II + 2 * HH) * 2);  // 256x1536
    _Float16* Xc16 = (_Float16*)take((size_t)BB * (II + HH) * 2);      // 256x1024
    _Float16* X216 = (_Float16*)take((size_t)BB * (II + HH) * 2);
    _Float16* Hn16 = (_Float16*)take((size_t)BB * HH * 2);
    float* r_ws   = (float*)take((size_t)BB * HH * 4);
    float* z_ws   = (float*)take((size_t)BB * HH * 4);
    float* sec_ws = (float*)take((size_t)BB * HH * 4);
    float* s_ws   = (float*)take((size_t)4 * BH * 4);                  // 4 partial sums

    // 1) weight conversions f32 -> f16
    convert_f32_f16_v4<<<(512 * 1536 / 4 + 255) / 256, 256, 0, stream>>>(Wd, Wd16, 512 * 1536 / 4);
    convert_f32_f16_v4<<<(512 * 1024 / 4 + 255) / 256, 256, 0, stream>>>(Wr, Wr16, 512 * 1024 / 4);
    convert_f32_f16_v4<<<(512 * 1024 / 4 + 255) / 256, 256, 0, stream>>>(Wz, Wz16, 512 * 1024 / 4);
    convert_f32_f16_v4<<<(512 * 1024 / 4 + 255) / 256, 256, 0, stream>>>(Wh, Wh16, 512 * 1024 / 4);
    convert_f32_f16_v4<<<(512 * 512 / 4 + 255) / 256, 256, 0, stream>>>(Wo, Wo16, 512 * 512 / 4);

    // 2) pack activations
    pack_inputs<<<dim3(6, 256), 256, 0, stream>>>(sample, hidden, d0, Xd16, Xc16);

    // 3) d_values = sigmoid(Xd @ Wd^T + bd) * 0.5
    gemm_wmma<0><<<64, 256, 0, stream>>>(Xd16, Wd16, bd, 1536, out_dvals, nullptr);

    // 4) filter partials + fused h_c_1 shift (dominant kernel: ~268 MB HBM traffic)
    filter_partial<<<dim3(BH / 4 / 256, 4), 256, 0, stream>>>(htens, out_dvals, s_ws, out_hc1);

    // 5/6) r, z gates
    gemm_wmma<1><<<64, 256, 0, stream>>>(Xc16, Wr16, br, 1024, r_ws, nullptr);
    gemm_wmma<1><<<64, 256, 0, stream>>>(Xc16, Wz16, bz, 1024, z_ws, nullptr);
    // 7) combined2 = [sample | r*hidden]
    pack_combined2<<<dim3(4, 256), 256, 0, stream>>>(sample, hidden, r_ws, X216);
    // 8) second = tanh(X2 @ Wh^T + bh) * (1 - z)
    gemm_wmma<2><<<64, 256, 0, stream>>>(X216, Wh16, bh, 1024, sec_ws, z_ws);
    // 9) hidden_new = second - sum(partials); also h_c_1 last slice + f16 copy
    combine_hidden<<<BH / 4 / 256, 256, 0, stream>>>(
        s_ws, sec_ws, out_hidden, out_hc1 + (size_t)(KK - 1) * BH, Hn16);
    // 10) output = hidden_new @ Wo^T + bo
    gemm_wmma<3><<<64, 256, 0, stream>>>(Hn16, Wo16, bo, 512, out_output, nullptr);
}
